// MambaMHISTModel_77446850282076
// MI455X (gfx1250) — matrix operand order repaired and checked by reading, not compiled, not run on hardware
//
#include <hip/hip_runtime.h>
#include <hip/hip_bf16.h>
#include <cstdint>

typedef _Float16 v16h __attribute__((ext_vector_type(16)));
typedef _Float16 h2   __attribute__((ext_vector_type(2)));
typedef float    v8f  __attribute__((ext_vector_type(8)));

#define BATCH   32
#define SEQ     1024
#define D_MODEL 512
#define NCLS    2
#define D_INNER 1024
#define D_STATE 64
#define DCONV   4
#define DT_RANK 32

#define TILE_M 128
#define TILE_N 128
#define TILE_K 32
#define LDS_STRIDE (TILE_K + 2)   // 34 halves = 17 dwords (odd) -> conflict-free, rows 4B aligned

// ---------------------------------------------------------------------------
// Generic f32-in/f32-out GEMM via v_wmma_f32_16x16x32_f16.
// C[M,N] = A[M,K](lda) @ B[K,N](ldb).
// Requirements (all call sites satisfy them):
//   - K multiple of 32, M multiple of 128, N even, lda/ldb even.
//   - RAGGED_N=false additionally requires N multiple of TILE_N.
// Block: 256 threads = 8 waves (4x2), block tile 128x128, wave tile 32x64.
// Register double-buffer: global loads for slab k+1 overlap WMMAs of slab k.
// ---------------------------------------------------------------------------
template <bool RAGGED_N>
__global__ __launch_bounds__(256)
void gemm_wmma_f16(const float* __restrict__ A, int lda,
                   const float* __restrict__ Bw, int ldb,
                   float* __restrict__ C, int ldc,
                   int M, int N, int K)
{
    __shared__ _Float16 As[TILE_M * LDS_STRIDE];   // [m][k]
    __shared__ _Float16 Bs[TILE_N * LDS_STRIDE];   // [n][k] (transposed on fill)

    const int tid  = threadIdx.x;
    const int wid  = tid >> 5;
    const int lane = tid & 31;
    const int wm   = wid & 3;        // 4 wave-rows of 32
    const int wn   = wid >> 2;       // 2 wave-cols of 64
    const int bm   = blockIdx.y * TILE_M;
    const int bn   = blockIdx.x * TILE_N;
    const int lrow = lane & 15;
    const int lhi  = lane >> 4;

    // Per-thread staging coordinates (8 float2 each for A and B per K-slab).
    const int am0 = tid >> 4;            // A row base (+16 per j), col pair = ac2
    const int ac2 = tid & 15;            // A col pair index (k = 2*ac2)
    const int bk0 = tid >> 6;            // B k-row base (+4 per j)
    const int bn2 = tid & 63;            // B col pair index (n = 2*bn2)
    const int gn  = bn + 2 * bn2;
    const int cn  = RAGGED_N ? ((gn <= N - 2) ? gn : (N - 2)) : gn;  // clamp (N even)

    v8f zero = {};
    v8f acc[2][4];
#pragma unroll
    for (int mi = 0; mi < 2; ++mi)
#pragma unroll
        for (int ni = 0; ni < 4; ++ni) acc[mi][ni] = zero;

    float2 ar[8], br[8];

    // Prologue: load slab 0 (unconditional, coalesced 64-bit loads).
#pragma unroll
    for (int j = 0; j < 8; ++j) {
        ar[j] = *(const float2*)&A[(size_t)(bm + am0 + 16 * j) * lda + 2 * ac2];
        br[j] = *(const float2*)&Bw[(size_t)(bk0 + 4 * j) * ldb + cn];
    }

    for (int k0 = 0; k0 < K; k0 += TILE_K) {
        // Drain in-flight loads into LDS (f32 -> f16).
#pragma unroll
        for (int j = 0; j < 8; ++j) {
            h2 ha; ha[0] = (_Float16)ar[j].x; ha[1] = (_Float16)ar[j].y;
            *(h2*)&As[(am0 + 16 * j) * LDS_STRIDE + 2 * ac2] = ha;
            float bx = br[j].x, by = br[j].y;
            if (RAGGED_N) {                      // branchless select, only when needed
                bx = (gn     < N) ? bx : 0.f;
                by = (gn + 1 < N) ? by : 0.f;
            }
            Bs[(2 * bn2    ) * LDS_STRIDE + bk0 + 4 * j] = (_Float16)bx;
            Bs[(2 * bn2 + 1) * LDS_STRIDE + bk0 + 4 * j] = (_Float16)by;
        }
        __syncthreads();

        // Kick off next slab's global loads; they complete under the WMMAs.
        if (k0 + TILE_K < K) {
            const int kn = k0 + TILE_K;
#pragma unroll
            for (int j = 0; j < 8; ++j) {
                ar[j] = *(const float2*)&A[(size_t)(bm + am0 + 16 * j) * lda + kn + 2 * ac2];
                br[j] = *(const float2*)&Bw[(size_t)(kn + bk0 + 4 * j) * ldb + cn];
            }
        }

        // A fragments (ISA 16-bit A 16x32 layout), 2 M-tiles per wave.
        v16h afrag[2];
#pragma unroll
        for (int mi = 0; mi < 2; ++mi) {
            h2* ap = (h2*)&afrag[mi];
            const int m = wm * 32 + mi * 16 + lrow;
#pragma unroll
            for (int v = 0; v < 8; ++v) {
                const int g  = v >> 2;
                const int kk = g * 16 + lhi * 8 + (v & 3) * 2;
                ap[v] = *(const h2*)&As[m * LDS_STRIDE + kk];
            }
        }
        // 4 B fragments x 2 A fragments = 8 WMMAs per slab per wave.
#pragma unroll
        for (int ni = 0; ni < 4; ++ni) {
            v16h bfrag;
            h2* bp = (h2*)&bfrag;
            const int n = wn * 64 + ni * 16 + lrow;
#pragma unroll
            for (int v = 0; v < 8; ++v) {
                const int kk = (v >> 2) * 16 + lhi * 8 + (v & 3) * 2;
                bp[v] = *(const h2*)&Bs[n * LDS_STRIDE + kk];
            }
#pragma unroll
            for (int mi = 0; mi < 2; ++mi)
                acc[mi][ni] = __builtin_amdgcn_wmma_f32_16x16x32_f16(
                    false, afrag[mi], false, bfrag, (short)0, acc[mi][ni], false, false);
        }
        __syncthreads();
    }

    // Store: D layout — vgpr r holds M = lhi*8 + r, N = lane&15.
#pragma unroll
    for (int ni = 0; ni < 4; ++ni) {
        const int n = bn + wn * 64 + ni * 16 + lrow;
        if (RAGGED_N && n >= N) continue;
#pragma unroll
        for (int mi = 0; mi < 2; ++mi) {
#pragma unroll
            for (int r = 0; r < 8; ++r) {
                const int m = bm + wm * 32 + mi * 16 + lhi * 8 + r;
                C[(size_t)m * ldc + n] = acc[mi][ni][r];
            }
        }
    }
}

// ---------------------------------------------------------------------------
// Depthwise causal conv (k=4) + SiLU.  xin = xz[..., :1024].
// ---------------------------------------------------------------------------
__global__ __launch_bounds__(256)
void conv_silu_kernel(const float* __restrict__ xz, const float* __restrict__ conv_w,
                      const float* __restrict__ conv_b, float* __restrict__ xc)
{
    const size_t idx = (size_t)blockIdx.x * 256 + threadIdx.x;
    if (idx >= (size_t)BATCH * SEQ * D_INNER) return;
    const int d = (int)(idx & (D_INNER - 1));
    const int l = (int)((idx >> 10) & (SEQ - 1));
    const int b = (int)(idx >> 20);
    float acc = conv_b[d];
#pragma unroll
    for (int k = 0; k < DCONV; ++k) {
        const int lp = l + k - (DCONV - 1);
        if (lp >= 0)
            acc += conv_w[k * D_INNER + d] * xz[(((size_t)b * SEQ + lp) << 11) + d];
    }
    xc[idx] = acc / (1.f + __expf(-acc));
}

// dt = softplus(dt_lin + b_dt), in place.
__global__ __launch_bounds__(256)
void softplus_kernel(float* __restrict__ dt, const float* __restrict__ b_dt)
{
    const size_t idx = (size_t)blockIdx.x * 256 + threadIdx.x;
    if (idx >= (size_t)BATCH * SEQ * D_INNER) return;
    const int d = (int)(idx & (D_INNER - 1));
    const float xv = dt[idx] + b_dt[d];
    dt[idx] = (xv > 20.f) ? xv : log1pf(__expf(xv));
}

// ---------------------------------------------------------------------------
// Selective scan: one thread per (b,d), 64-wide state in registers.
// B_t/C_t staged in LDS for SCHUNK timesteps at a time (2 barriers per chunk
// instead of per step); per-thread u/dt/z streams batched per chunk so the
// exp/FMA chain runs with all chunk loads already in flight.
// Fuses +xc*Dp and *silu(z).
// ---------------------------------------------------------------------------
#define SCHUNK 8
__global__ __launch_bounds__(256)
void scan_kernel(const float* __restrict__ xc, const float* __restrict__ dt,
                 const float* __restrict__ xdbc, const float* __restrict__ A_log,
                 const float* __restrict__ Dp, const float* __restrict__ xz,
                 float* __restrict__ yout)
{
    const int b = blockIdx.x;
    const int d = blockIdx.y * 256 + threadIdx.x;

    float Arow[D_STATE];
#pragma unroll
    for (int n = 0; n < D_STATE; ++n)
        Arow[n] = -__expf(A_log[(size_t)d * D_STATE + n]);
    float h[D_STATE];
#pragma unroll
    for (int n = 0; n < D_STATE; ++n) h[n] = 0.f;
    const float Dd = Dp[d];

    __shared__ float BC[SCHUNK][2 * D_STATE];   // [t][0:64]=B_t, [t][64:128]=C_t

    for (int l0 = 0; l0 < SEQ; l0 += SCHUNK) {
        const size_t row0 = (size_t)b * SEQ + l0;

        __syncthreads();   // previous chunk fully consumed
#pragma unroll
        for (int j = 0; j < (SCHUNK * 2 * D_STATE) / 256; ++j) {   // 4 coalesced loads
            const int i = threadIdx.x + j * 256;
            const int t = i >> 7;
            const int n = i & 127;
            BC[t][n] = xdbc[(row0 + t) * 160 + DT_RANK + n];
        }
        __syncthreads();

        // Batch the per-thread streams for the whole chunk.
        float uv[SCHUNK], dtv[SCHUNK], zvv[SCHUNK];
#pragma unroll
        for (int t = 0; t < SCHUNK; ++t) {
            const size_t row = row0 + t;
            uv[t]  = xc[(row << 10) + d];
            dtv[t] = dt[(row << 10) + d];
            zvv[t] = xz[(row << 11) + D_INNER + d];
        }

#pragma unroll 1   // keep code size sane; state loop below is fully unrolled
        for (int t = 0; t < SCHUNK; ++t) {
            const float du = dtv[t] * uv[t];
            float y = 0.f;
#pragma unroll
            for (int n = 0; n < D_STATE; ++n) {
                const float dA = __expf(dtv[t] * Arow[n]);
                h[n] = h[n] * dA + du * BC[t][n];
                y = fmaf(h[n], BC[t][D_STATE + n], y);
            }
            const float zv   = zvv[t];
            const float silz = zv / (1.f + __expf(-zv));
            yout[((row0 + t) << 10) + d] = (y + uv[t] * Dd) * silz;
        }
    }
}

// pooled[b,dm] = mean_l hmod[b,l,dm]
__global__ __launch_bounds__(256)
void pool_kernel(const float* __restrict__ hmod, float* __restrict__ pooled)
{
    const int b  = blockIdx.y;
    const int dm = blockIdx.x * 256 + threadIdx.x;
    float s = 0.f;
    for (int l = 0; l < SEQ; ++l)
        s += hmod[((size_t)b * SEQ + l) * D_MODEL + dm];
    pooled[b * D_MODEL + dm] = s * (1.0f / SEQ);
}

// out = pooled @ W_cls + b_cls  (32x512 @ 512x2 — tiny, scalar)
__global__ void cls_kernel(const float* __restrict__ pooled, const float* __restrict__ W_cls,
                           const float* __restrict__ b_cls, float* __restrict__ out)
{
    const int i = threadIdx.x;
    if (i >= BATCH * NCLS) return;
    const int b = i >> 1, c = i & 1;
    float s = b_cls[c];
    for (int k = 0; k < D_MODEL; ++k)
        s = fmaf(pooled[b * D_MODEL + k], W_cls[k * NCLS + c], s);
    out[i] = s;
}

// ---------------------------------------------------------------------------
extern "C" void kernel_launch(void* const* d_in, const int* in_sizes, int n_in,
                              void* d_out, int out_size, void* d_ws, size_t ws_size,
                              hipStream_t stream)
{
    (void)in_sizes; (void)n_in; (void)out_size; (void)ws_size;

    const float* x      = (const float*)d_in[0];
    const float* W_in   = (const float*)d_in[1];
    const float* conv_w = (const float*)d_in[2];
    const float* conv_b = (const float*)d_in[3];
    const float* W_x    = (const float*)d_in[4];
    const float* W_dt   = (const float*)d_in[5];
    const float* b_dt   = (const float*)d_in[6];
    const float* A_log  = (const float*)d_in[7];
    const float* Dp     = (const float*)d_in[8];
    const float* W_out  = (const float*)d_in[9];
    const float* W_cls  = (const float*)d_in[10];
    const float* b_cls  = (const float*)d_in[11];
    float* out = (float*)d_out;

    const int M = BATCH * SEQ;                 // 32768
    float* ws     = (float*)d_ws;
    float* xz     = ws;                        // M*2048
    float* xc     = xz   + (size_t)M * 2048;   // M*1024
    float* xdbc   = xc   + (size_t)M * 1024;   // M*160
    float* dt     = xdbc + (size_t)M * 160;    // M*1024
    float* yy     = dt   + (size_t)M * 1024;   // M*1024
    float* hmod   = yy   + (size_t)M * 1024;   // M*512
    float* pooled = hmod + (size_t)M * 512;    // 32*512

    dim3 blk(256);

    // 1) xz = x @ W_in   [32768 x 512 x 2048]  (N % 128 == 0 -> dense path)
    { dim3 g((2 * D_INNER) / TILE_N, M / TILE_M);
      gemm_wmma_f16<false><<<g, blk, 0, stream>>>(x, D_MODEL, W_in, 2 * D_INNER,
                                                  xz, 2 * D_INNER, M, 2 * D_INNER, D_MODEL); }
    // 2) xc = silu(causal_conv(xin))
    { size_t total = (size_t)M * D_INNER;
      conv_silu_kernel<<<dim3((unsigned)((total + 255) / 256)), blk, 0, stream>>>(xz, conv_w, conv_b, xc); }
    // 3) xdbc = xc @ W_x   [32768 x 1024 x 160]  (ragged N)
    { dim3 g((DT_RANK + 2 * D_STATE + TILE_N - 1) / TILE_N, M / TILE_M);
      gemm_wmma_f16<true><<<g, blk, 0, stream>>>(xc, D_INNER, W_x, DT_RANK + 2 * D_STATE,
                                                 xdbc, DT_RANK + 2 * D_STATE,
                                                 M, DT_RANK + 2 * D_STATE, D_INNER); }
    // 4) dt_lin = dt_raw @ W_dt   [32768 x 32 x 1024]  (A strided inside xdbc)
    { dim3 g(D_INNER / TILE_N, M / TILE_M);
      gemm_wmma_f16<false><<<g, blk, 0, stream>>>(xdbc, DT_RANK + 2 * D_STATE, W_dt, D_INNER,
                                                  dt, D_INNER, M, D_INNER, DT_RANK); }
    // 5) dt = softplus(dt_lin + b_dt)
    { size_t total = (size_t)M * D_INNER;
      softplus_kernel<<<dim3((unsigned)((total + 255) / 256)), blk, 0, stream>>>(dt, b_dt); }
    // 6) selective scan fused with +xc*Dp and *silu(z)
    { dim3 g(BATCH, D_INNER / 256);
      scan_kernel<<<g, blk, 0, stream>>>(xc, dt, xdbc, A_log, Dp, xz, yy); }
    // 7) hmod = yy @ W_out   [32768 x 1024 x 512]
    { dim3 g(D_MODEL / TILE_N, M / TILE_M);
      gemm_wmma_f16<false><<<g, blk, 0, stream>>>(yy, D_INNER, W_out, D_MODEL,
                                                  hmod, D_MODEL, M, D_MODEL, D_INNER); }
    // 8) pooled = mean over L
    { dim3 g(D_MODEL / 256, BATCH);
      pool_kernel<<<g, blk, 0, stream>>>(hmod, pooled); }
    // 9) out = pooled @ W_cls + b_cls
    cls_kernel<<<1, 64, 0, stream>>>(pooled, W_cls, b_cls, out);
}
